// SelfAttentionV4_13185549598786
// MI455X (gfx1250) — matrix operand order repaired and checked by reading, compiled, not run on hardware
//
#include <hip/hip_runtime.h>
#include <hip/hip_bf16.h>

typedef __attribute__((ext_vector_type(16))) _Float16 v16h;
typedef __attribute__((ext_vector_type(8)))  float    v8f;

#define B_SZ 4
#define T_SZ 2048
#define DM   1024
#define DK   512
#define DV   512

union Frag16 {
    v16h   v;
    float4 f[2];
    _Float16 h[16];
};

__device__ __forceinline__ void load_frag32(const _Float16* p, Frag16& f) {
    f.f[0] = *(const float4*)(p);
    f.f[1] = *(const float4*)(p + 8);
}

__device__ __forceinline__ float rowmax16(float x) {
    x = fmaxf(x, __shfl_xor(x, 1, 32));
    x = fmaxf(x, __shfl_xor(x, 2, 32));
    x = fmaxf(x, __shfl_xor(x, 4, 32));
    x = fmaxf(x, __shfl_xor(x, 8, 32));
    return x;
}
__device__ __forceinline__ float rowsum16(float x) {
    x += __shfl_xor(x, 1, 32);
    x += __shfl_xor(x, 2, 32);
    x += __shfl_xor(x, 4, 32);
    x += __shfl_xor(x, 8, 32);
    return x;
}

// ---------------- Stage 1: weight transpose + f32->f16 convert ----------------
// W: [DM][DK] f32 row-major  ->  Wt: [DK][DM] f16 row-major
__global__ void prep_weights_kernel(const float* __restrict__ W,
                                    _Float16* __restrict__ Wt) {
    int idx = blockIdx.x * blockDim.x + threadIdx.x;   // idx = d*DK + n
    if (idx >= DM * DK) return;
    int d = idx / DK;
    int n = idx % DK;
    Wt[(size_t)n * DM + d] = (_Float16)W[idx];
}

// ---------------- Stage 2: projection GEMM  X[8192x1024] * Wt^T -> [8192x512] f16
// block = 256 threads (8 waves). wave w handles rows r0..r0+15, cols n0..n0+63.
__global__ __launch_bounds__(256) void proj_kernel(
    const float*    __restrict__ X,
    const _Float16* __restrict__ Wt,
    const float*    __restrict__ bias,
    _Float16*       __restrict__ Out,
    int transposed)
{
    const int lane = threadIdx.x & 31;
    const int wave = threadIdx.x >> 5;
    const int l15  = lane & 15;
    const int hi   = lane >> 4;
    const int r0   = blockIdx.x * 128 + wave * 16;   // global row base
    const int n0   = blockIdx.y * 64;                // output col base

    v8f acc[4];
#pragma unroll
    for (int nt = 0; nt < 4; ++nt) { v8f z = {}; acc[nt] = z; }

    const float* xrow = X + (size_t)(r0 + l15) * DM;

    // double-buffered raw A rows (f32)
    float g0[2][8], g1[2][8];
    {
        *(float4*)&g0[0][0] = *(const float4*)(xrow + 8 * hi);
        *(float4*)&g0[0][4] = *(const float4*)(xrow + 8 * hi + 4);
        *(float4*)&g1[0][0] = *(const float4*)(xrow + 16 + 8 * hi);
        *(float4*)&g1[0][4] = *(const float4*)(xrow + 16 + 8 * hi + 4);
    }

#pragma unroll 2
    for (int kt = 0; kt < DM / 32; ++kt) {
        const int cur = kt & 1;
        const int nxt = cur ^ 1;
        if (kt < DM / 32 - 1) {
            const int k0n = (kt + 1) * 32;
            *(float4*)&g0[nxt][0] = *(const float4*)(xrow + k0n + 8 * hi);
            *(float4*)&g0[nxt][4] = *(const float4*)(xrow + k0n + 8 * hi + 4);
            *(float4*)&g1[nxt][0] = *(const float4*)(xrow + k0n + 16 + 8 * hi);
            *(float4*)&g1[nxt][4] = *(const float4*)(xrow + k0n + 16 + 8 * hi + 4);
        }
        const int k0 = kt * 32;
        // A fragment: lane row = l15, halfs h -> K = h + 8*((h>=8) + hi)
        Frag16 a;
#pragma unroll
        for (int i = 0; i < 8; ++i) {
            a.h[i]     = (_Float16)g0[cur][i];
            a.h[8 + i] = (_Float16)g1[cur][i];
        }
        // load all 4 B fragments first, then issue the 4 WMMAs
        Frag16 bf[4];
#pragma unroll
        for (int nt = 0; nt < 4; ++nt) {
            const _Float16* wrow =
                Wt + (size_t)(n0 + nt * 16 + l15) * DM + k0 + hi * 8;
            bf[nt].f[0] = *(const float4*)(wrow);
            bf[nt].f[1] = *(const float4*)(wrow + 16);
        }
#pragma unroll
        for (int nt = 0; nt < 4; ++nt) {
            acc[nt] = __builtin_amdgcn_wmma_f32_16x16x32_f16(
                false, a.v, false, bf[nt].v, (short)0, acc[nt], false, false);
        }
    }

#pragma unroll
    for (int nt = 0; nt < 4; ++nt) {
        const int   n  = n0 + nt * 16 + l15;
        const float bv = bias[n];
#pragma unroll
        for (int i = 0; i < 8; ++i) {
            const int   r   = r0 + i + 8 * hi;        // global row (b*T + t)
            const float val = acc[nt][i] + bv;
            if (!transposed) {
                Out[(size_t)r * DK + n] = (_Float16)val;
            } else {
                const int bb = r >> 11;               // r / T
                const int t  = r & (T_SZ - 1);        // r % T
                Out[((size_t)(bb * DV + n)) * T_SZ + t] = (_Float16)val;
            }
        }
    }
}

// ---------------- Stage 3: causal flash attention ----------------
// block = 128 threads (4 waves); wave handles 16 query rows x full 512 v-cols.
__global__ __launch_bounds__(128) void attn_kernel(
    const _Float16* __restrict__ Qh,   // [B*T][512] f16
    const _Float16* __restrict__ Kh,   // [B*T][512] f16
    const _Float16* __restrict__ Vt,   // [B][512][T] f16 (transposed)
    float*          __restrict__ Out)  // [B*T][512] f32
{
    __shared__ _Float16 Plds[4][16][32];

    const int lane = threadIdx.x & 31;
    const int wave = threadIdx.x >> 5;
    const int l15  = lane & 15;
    const int hi   = lane >> 4;
    const int b    = blockIdx.y;
    const int t0   = blockIdx.x * 64 + wave * 16;     // query row base in batch

    v8f acc[32];
#pragma unroll
    for (int nt = 0; nt < 32; ++nt) { v8f z = {}; acc[nt] = z; }

    float m_i[8], l_i[8];
#pragma unroll
    for (int i = 0; i < 8; ++i) { m_i[i] = -1.0e30f; l_i[i] = 0.0f; }

    const float scale = 0.04419417382415922f;         // 1/sqrt(512)
    const float L2E   = 1.4426950408889634f;
    const _Float16* qbase = Qh + ((size_t)(b * T_SZ + t0 + l15)) * DK;
    const _Float16* vbase = Vt + ((size_t)(b * DV + l15)) * T_SZ;
    const int nchunk = (t0 + 15) / 32 + 1;            // causal bound, 32 keys/chunk

    for (int ch = 0; ch < nchunk; ++ch) {
        const int s0 = ch * 32;

        // ---- prefetch next chunk's K rows and V tile (global_prefetch_b8) ----
        if (ch + 1 < nchunk) {
            const _Float16* nk =
                Kh + ((size_t)(b * T_SZ + s0 + 32 + lane)) * DK;   // 32 rows, 1 KB each
            __builtin_prefetch(nk, 0, 3);
            __builtin_prefetch(nk + 128, 0, 3);
            __builtin_prefetch(nk + 256, 0, 3);
            __builtin_prefetch(nk + 384, 0, 3);
            const _Float16* nv =
                Vt + ((size_t)(b * DV + lane)) * T_SZ + s0 + 32;   // 512 rows, 64 B each
#pragma unroll
            for (int i = 0; i < 16; ++i)
                __builtin_prefetch(nv + (size_t)i * 32 * T_SZ, 0, 3);
        }

        // ---- scores: two 16x16 tiles over full d_k = 512, double-buffered ----
        v8f sc0 = {}, sc1 = {};
        const _Float16* kb0 =
            Kh + ((size_t)(b * T_SZ + s0 + l15)) * DK + hi * 16;
        const _Float16* kb1 =
            Kh + ((size_t)(b * T_SZ + s0 + 16 + l15)) * DK + hi * 16;

        Frag16 qa[2], kfa[2], kfb[2];
        qa[0].f[0] = *(const float4*)(qbase + 16 * hi);
        qa[0].f[1] = *(const float4*)(qbase + 16 * hi + 8);
        load_frag32(kb0, kfa[0]);
        load_frag32(kb1, kfb[0]);
#pragma unroll
        for (int kt = 0; kt < 16; ++kt) {
            const int cur = kt & 1;
            const int nxt = cur ^ 1;
            if (kt < 15) {
                const int k0n = (kt + 1) * 32;
                qa[nxt].f[0] = *(const float4*)(qbase + k0n + 16 * hi);
                qa[nxt].f[1] = *(const float4*)(qbase + k0n + 16 * hi + 8);
                load_frag32(kb0 + k0n, kfa[nxt]);
                load_frag32(kb1 + k0n, kfb[nxt]);
            }
            sc0 = __builtin_amdgcn_wmma_f32_16x16x32_f16(
                false, qa[cur].v, false, kfa[cur].v, (short)0, sc0, false, false);
            sc1 = __builtin_amdgcn_wmma_f32_16x16x32_f16(
                false, qa[cur].v, false, kfb[cur].v, (short)0, sc1, false, false);
        }

        // ---- online softmax (per row; row lives across 16 lanes of one half) ----
        float alpha[8];
#pragma unroll
        for (int i = 0; i < 8; ++i) {
            const int tg  = t0 + i + 8 * hi;          // query row
            const int sg0 = s0 + l15;                 // key col (tile 0)
            const int sg1 = s0 + 16 + l15;            // key col (tile 1)
            float s0v = sc0[i] * scale;
            float s1v = sc1[i] * scale;
            if (sg0 > tg) s0v = -1.0e30f;
            if (sg1 > tg) s1v = -1.0e30f;
            const float rmax = rowmax16(fmaxf(s0v, s1v));
            const float mnew = fmaxf(m_i[i], rmax);
            alpha[i] = exp2f((m_i[i] - mnew) * L2E);
            const float p0 = exp2f((s0v - mnew) * L2E);
            const float p1 = exp2f((s1v - mnew) * L2E);
            const float rsum = rowsum16(p0 + p1);
            l_i[i] = l_i[i] * alpha[i] + rsum;
            m_i[i] = mnew;
            // write P in C-layout position; re-read below in A-layout
            Plds[wave][i + 8 * hi][l15]      = (_Float16)p0;
            Plds[wave][i + 8 * hi][16 + l15] = (_Float16)p1;
        }
        asm volatile("s_wait_dscnt 0" ::: "memory");

        // ---- rescale accumulators ----
#pragma unroll
        for (int nt = 0; nt < 32; ++nt) {
#pragma unroll
            for (int i = 0; i < 8; ++i) acc[nt][i] *= alpha[i];
        }

        // ---- P as A-fragment (16 rows x 32 keys) ----
        Frag16 pf;
        pf.f[0] = *(const float4*)&Plds[wave][l15][16 * hi];
        pf.f[1] = *(const float4*)&Plds[wave][l15][16 * hi + 8];

        // ---- AV: 32 n-tiles, one K=32 WMMA each, double-buffered V frags ----
        Frag16 vf[2];
        load_frag32(vbase + s0 + hi * 16, vf[0]);
#pragma unroll
        for (int nt = 0; nt < 32; ++nt) {
            const int cur = nt & 1;
            const int nxt = cur ^ 1;
            if (nt < 31) {
                load_frag32(vbase + (size_t)(nt + 1) * 16 * T_SZ + s0 + hi * 16,
                            vf[nxt]);
            }
            acc[nt] = __builtin_amdgcn_wmma_f32_16x16x32_f16(
                false, pf.v, false, vf[cur].v, (short)0, acc[nt], false, false);
        }
    }

    // ---- epilogue: normalize and store f32 ----
#pragma unroll
    for (int i = 0; i < 8; ++i) {
        const float inv = 1.0f / l_i[i];
        const int   r   = b * T_SZ + t0 + i + 8 * hi;
        float* orow = Out + (size_t)r * DV;
#pragma unroll
        for (int nt = 0; nt < 32; ++nt) {
            orow[nt * 16 + l15] = acc[nt][i] * inv;
        }
    }
}

extern "C" void kernel_launch(void* const* d_in, const int* in_sizes, int n_in,
                              void* d_out, int out_size, void* d_ws, size_t ws_size,
                              hipStream_t stream) {
    const float* Xq = (const float*)d_in[0];
    const float* Xk = (const float*)d_in[1];
    const float* Xv = (const float*)d_in[2];
    const float* Wq = (const float*)d_in[3];
    const float* bq = (const float*)d_in[4];
    const float* Wk = (const float*)d_in[5];
    const float* bk = (const float*)d_in[6];
    const float* Wv = (const float*)d_in[7];
    const float* bv = (const float*)d_in[8];
    float* Out = (float*)d_out;

    // workspace carve-up (f16): 3x 0.5M weights + 3x 4M activations = ~27 MB
    _Float16* WtQ = (_Float16*)d_ws;
    _Float16* WtK = WtQ + (size_t)DK * DM;
    _Float16* WtV = WtK + (size_t)DK * DM;
    _Float16* Qh  = WtV + (size_t)DK * DM;
    _Float16* Kh  = Qh  + (size_t)B_SZ * T_SZ * DK;
    _Float16* Vt  = Kh  + (size_t)B_SZ * T_SZ * DK;

    const int nw = DM * DK;
    prep_weights_kernel<<<(nw + 255) / 256, 256, 0, stream>>>(Wq, WtQ);
    prep_weights_kernel<<<(nw + 255) / 256, 256, 0, stream>>>(Wk, WtK);
    prep_weights_kernel<<<(nw + 255) / 256, 256, 0, stream>>>(Wv, WtV);

    dim3 pg(64, 8);   // 8192 rows / 128, 512 cols / 64
    proj_kernel<<<pg, 256, 0, stream>>>(Xq, WtQ, bq, Qh, 0);
    proj_kernel<<<pg, 256, 0, stream>>>(Xk, WtK, bk, Kh, 0);
    proj_kernel<<<pg, 256, 0, stream>>>(Xv, WtV, bv, Vt, 1);

    dim3 ag(T_SZ / 64, B_SZ);   // (32, 4)
    attn_kernel<<<ag, 128, 0, stream>>>(Qh, Kh, Vt, Out);
}
